// myPCANormSVDPI_8151847928251
// MI455X (gfx1250) — matrix operand order repaired and checked by reading, not compiled, not run on hardware
//
#include <hip/hip_runtime.h>

#define C_DIM 256
#define N_IMG 32
#define HW_DIM 4096
#define M_TOT (N_IMG * HW_DIM) /* 131072 */
#define EPSF 1e-4f
#define NEIG 16
#define NPI 24

#define GK  256   // K-depth staged per TDM stage in gram kernel
#define GLP 264   // padded LDS row: 256 data + 8 pad bf16 (16B pad)
#define PLP 72    // padded LDS row: 64 data + 8 pad bf16

typedef __attribute__((ext_vector_type(16))) __bf16 v16bf;
typedef __attribute__((ext_vector_type(8)))  float  v8f;
typedef __attribute__((ext_vector_type(4)))  unsigned int v4u;
typedef __attribute__((ext_vector_type(8)))  int v8i;
typedef __attribute__((ext_vector_type(4)))  int v4i;

__device__ __forceinline__ unsigned lds_off(const void* p) {
    // generic pointer to LDS: low 32 bits are the LDS byte offset (ISA 10.2)
    return (unsigned)(unsigned long long)p;
}

// ---------------------------------------------------------------------------
// Tensor Data Mover: 2-D tile (tile_k elems per row x rows) of bf16 from
// global (row stride strideElems) into LDS at ldsA, with per-row LDS padding.
// D# layout per CDNA5 ISA ch.8: group0 = count/lds/global/type, group1 = dims.
// padIntCode: pad after 2^(c+1) DWORDs; padAmtCode: pad (c+1) DWORDs.
// ---------------------------------------------------------------------------
__device__ __forceinline__ void tdm_load_2d(unsigned ldsA, const __bf16* g,
                                            unsigned tile_k, unsigned rows,
                                            unsigned long long strideElems,
                                            unsigned padIntCode, unsigned padAmtCode) {
    unsigned long long ga = (unsigned long long)g;
    v4u g0;
    g0.x = 1u;                                           // count=1, user mode
    g0.y = ldsA;                                         // lds_addr
    g0.z = (unsigned)ga;                                 // global_addr[31:0]
    g0.w = ((unsigned)(ga >> 32) & 0x01FFFFFFu)          // global_addr[56:32]
         | 0x80000000u;                                  // type=2 ("image")
    v8i g1;
    g1[0] = (int)((1u << 16)                             // data_size = 2 bytes
                | (1u << 20)                             // pad_enable
                | (padIntCode << 22) | (padAmtCode << 25));
    g1[1] = 0;                                           // tensor_dim0 = 1<<30 (lo16=0)
    g1[2] = (int)0x4000u;                                // tensor_dim0 hi16
    g1[3] = (int)(0x4000u | (tile_k << 16));             // tensor_dim1 hi16 | tile_dim0
    g1[4] = (int)rows;                                   // tile_dim1 (tile_dim2=0)
    g1[5] = (int)(unsigned)(strideElems & 0xFFFFFFFFull);// tensor_dim0_stride lo
    g1[6] = (int)(unsigned)(strideElems >> 32);          // stride hi (dim1_stride=0)
    g1[7] = 0;
    v4i z4 = {0, 0, 0, 0};
#if defined(__clang_major__) && __clang_major__ >= 23
    v8i z8 = {0, 0, 0, 0, 0, 0, 0, 0};
    __builtin_amdgcn_tensor_load_to_lds(g0, g1, z4, z4, z8, 0);
#else
    __builtin_amdgcn_tensor_load_to_lds(g0, g1, z4, z4, 0);
#endif
}

// ---------------------------------------------------------------------------
// K1: per-channel mean / rstd over m = N*H*W samples. One block per channel.
// ---------------------------------------------------------------------------
__global__ __launch_bounds__(256) void stats_kernel(const float* __restrict__ x,
                                                    float* __restrict__ mu,
                                                    float* __restrict__ rstd) {
    __shared__ float s1[256], s2[256];
    const int c = blockIdx.x, t = threadIdx.x;
    float sum = 0.f, sq = 0.f;
    for (int n = 0; n < N_IMG; ++n) {
        const float4* p = (const float4*)(x + ((size_t)(n * C_DIM + c)) * HW_DIM) + t * 4;
#pragma unroll
        for (int u = 0; u < 4; ++u) {
            float4 v = p[u];
            sum += v.x + v.y + v.z + v.w;
            sq  += v.x * v.x + v.y * v.y + v.z * v.z + v.w * v.w;
        }
    }
    s1[t] = sum; s2[t] = sq;
    __syncthreads();
    for (int s = 128; s > 0; s >>= 1) {
        if (t < s) { s1[t] += s1[t + s]; s2[t] += s2[t + s]; }
        __syncthreads();
    }
    if (t == 0) {
        const float m = (float)M_TOT;
        const float mean = s1[0] / m;
        const float var  = (s2[0] - s1[0] * s1[0] / m) / (m - 1.0f); // unbiased
        mu[c]   = mean;
        rstd[c] = rsqrtf(var + EPSF);
    }
}

// ---------------------------------------------------------------------------
// K1b: materialize xb = bf16((x - mu) * rstd) as (C, N*H*W) row-major.
// grid = (C, N), fully coalesced loads and stores.
// ---------------------------------------------------------------------------
__global__ __launch_bounds__(256) void normconv_kernel(const float* __restrict__ x,
                                                       const float* __restrict__ mu,
                                                       const float* __restrict__ rstd,
                                                       __bf16* __restrict__ xb) {
    const int c = blockIdx.x, n = blockIdx.y, t = threadIdx.x;
    const float muc = mu[c], rsc = rstd[c];
    const float4* src = (const float4*)(x + ((size_t)(n * C_DIM + c)) * HW_DIM);
    __bf16* dst = xb + (size_t)c * M_TOT + (size_t)n * HW_DIM;
#pragma unroll
    for (int p = 0; p < 4; ++p) {
        float4 v = src[p * 256 + t];
        const int o = (p * 256 + t) * 4;
        dst[o + 0] = (__bf16)((v.x - muc) * rsc);
        dst[o + 1] = (__bf16)((v.y - muc) * rsc);
        dst[o + 2] = (__bf16)((v.z - muc) * rsc);
        dst[o + 3] = (__bf16)((v.w - muc) * rsc);
    }
}

// ---------------------------------------------------------------------------
// K2: Gram matrix via bf16 WMMA with double-buffered TDM staging.
// grid = (16 tile pairs of 64x64, 32 K-chunks of 4096), block = 256 (8 waves).
// Wave 0 drives the Tensor Data Mover; all waves compute; TENSORcnt + barrier
// rotate the two LDS buffers.
// ---------------------------------------------------------------------------
__global__ __launch_bounds__(256) void gram_kernel(const __bf16* __restrict__ xb,
                                                   float* __restrict__ xxt) {
    extern __shared__ __bf16 gsm[];
    __bf16 (*Ab)[64][GLP] = (__bf16(*)[64][GLP])gsm;                 // [2][64][GLP]
    __bf16 (*Bb)[64][GLP] = (__bf16(*)[64][GLP])(gsm + 2 * 64 * GLP);

    const int t = threadIdx.x, lane = t & 31, w = t >> 5;
    const int rbase = (blockIdx.x >> 2) * 64;
    const int cbase = (blockIdx.x & 3) * 64;
    const int n     = blockIdx.y;
    const int mgrp = (w >> 2) * 32, ngrp = (w & 3) * 16;
    const int rl = lane & 15;
    const int ab = (lane & 16) ? 8 : 0;      // A-frag K base (ISA 7.12.2)
    const int bb = (lane & 16) ? 16 : 0;     // unused; B-frag also uses ab

    const __bf16* gA = xb + (size_t)rbase * M_TOT + (size_t)n * HW_DIM;
    const __bf16* gB = xb + (size_t)cbase * M_TOT + (size_t)n * HW_DIM;

    v8f acc0 = {}, acc1 = {};

    if (t < 32) {                                     // prologue: fill buffer 0
        tdm_load_2d(lds_off(&Ab[0][0][0]), gA, GK, 64, M_TOT, 6, 3);
        tdm_load_2d(lds_off(&Bb[0][0][0]), gB, GK, 64, M_TOT, 6, 3);
        __builtin_amdgcn_s_wait_tensorcnt(0);
    }
    __syncthreads();

    const int NIT = HW_DIM / GK;                      // 16 stages
    for (int it = 0; it < NIT; ++it) {
        const int cur = it & 1;
        if (t < 32 && (it + 1) < NIT) {               // async-load next stage
            const int nxt = cur ^ 1;
            tdm_load_2d(lds_off(&Ab[nxt][0][0]), gA + (size_t)(it + 1) * GK,
                        GK, 64, M_TOT, 6, 3);
            tdm_load_2d(lds_off(&Bb[nxt][0][0]), gB + (size_t)(it + 1) * GK,
                        GK, 64, M_TOT, 6, 3);
        }
#pragma unroll
        for (int ks = 0; ks < GK; ks += 32) {
            v16bf a0, a1, b0;
#pragma unroll
            for (int j = 0; j < 8; ++j) {
                a0[j]     = Ab[cur][mgrp + rl][ks + ab + j];
                a0[j + 8] = Ab[cur][mgrp + rl][ks + 16 + ab + j];
                a1[j]     = Ab[cur][mgrp + 16 + rl][ks + ab + j];
                a1[j + 8] = Ab[cur][mgrp + 16 + rl][ks + 16 + ab + j];
            }
#pragma unroll
            for (int j = 0; j < 8; ++j) {
                b0[j]     = Bb[cur][ngrp + rl][ks + ab + j];
                b0[j + 8] = Bb[cur][ngrp + rl][ks + 16 + ab + j];
            }

            acc0 = __builtin_amdgcn_wmma_f32_16x16x32_bf16(false, a0, false, b0,
                                                           (short)0, acc0, false, false);
            acc1 = __builtin_amdgcn_wmma_f32_16x16x32_bf16(false, a1, false, b0,
                                                           (short)0, acc1, false, false);
        }
        if (t < 32) __builtin_amdgcn_s_wait_tensorcnt(0);
        __syncthreads();
    }

    // C/D layout: VGPR i -> row (lane<16 ? i : i+8), col = lane%16
    const int rowoff = (lane & 16) ? 8 : 0;
    const int col    = cbase + ngrp + rl;
#pragma unroll
    for (int i = 0; i < 8; ++i) {
        atomicAdd(&xxt[(size_t)(rbase + mgrp + rowoff + i) * C_DIM + col], acc0[i]);
        atomicAdd(&xxt[(size_t)(rbase + mgrp + 16 + rowoff + i) * C_DIM + col], acc1[i]);
    }
}

// ---------------------------------------------------------------------------
// K3: deflated power iteration on A = xxt/m + eps*I, whole matrix in LDS
// (256 KB dynamic LDS; WGP has 320 KB). Emits Pw = diag(weight) * V * V^T (bf16).
// ---------------------------------------------------------------------------
__global__ __launch_bounds__(256) void eigen_kernel(const float* __restrict__ xxt,
                                                    const float* __restrict__ weight,
                                                    __bf16* __restrict__ Pw) {
    extern __shared__ float A[];                 // 256*256 floats
    __shared__ float v[256], av[256], red[256];
    __shared__ float Vm[256][NEIG];
    const int t = threadIdx.x;
    const float invm = 1.0f / (float)M_TOT;

    for (int j = 0; j < C_DIM; ++j)
        A[(size_t)t * C_DIM + j] = xxt[(size_t)t * C_DIM + j] * invm + ((j == t) ? EPSF : 0.0f);
    __syncthreads();

    for (int e = 0; e < NEIG; ++e) {
        unsigned h = (unsigned)(t * 2654435761u) ^ (unsigned)(e * 0x9E3779B9u);
        v[t] = ((float)((h >> 9) & 0xFFFF)) * (1.0f / 32768.0f) - 1.0f + 1e-3f;
        __syncthreads();

        for (int it = 0; it < NPI; ++it) {
            float s = 0.f;
            for (int j = 0; j < C_DIM; ++j) s += A[(size_t)t * C_DIM + j] * v[j];
            av[t] = s; red[t] = s * s;
            __syncthreads();
            for (int r = 128; r > 0; r >>= 1) {
                if (t < r) red[t] += red[t + r];
                __syncthreads();
            }
            const float rn = rsqrtf(red[0] + 1e-30f);
            __syncthreads();
            v[t] = av[t] * rn;
            __syncthreads();
        }
        Vm[t][e] = v[t];

        float wv = 0.f;                              // deflate A -= (Av)v^T
        for (int j = 0; j < C_DIM; ++j) wv += A[(size_t)t * C_DIM + j] * v[j];
        for (int j = 0; j < C_DIM; ++j) A[(size_t)t * C_DIM + j] -= wv * v[j];
        __syncthreads();
    }

    const float wt = weight[t];
    float vi[NEIG];
#pragma unroll
    for (int i = 0; i < NEIG; ++i) vi[i] = Vm[t][i];
    for (int c2 = 0; c2 < C_DIM; ++c2) {
        float s = 0.f;
#pragma unroll
        for (int i = 0; i < NEIG; ++i) s += vi[i] * Vm[c2][i];
        Pw[(size_t)t * C_DIM + c2] = (__bf16)(s * wt);
    }
}

// ---------------------------------------------------------------------------
// K4: out = Pw (256x256 bf16) * xn(bf16) + bias. One TDM load stages the
// 256x64 xn column block; 64 WMMAs per wave; fp32 coalesced stores.
// ---------------------------------------------------------------------------
__global__ __launch_bounds__(256) void proj_kernel(const __bf16* __restrict__ xb,
                                                   const __bf16* __restrict__ Pw,
                                                   const float* __restrict__ bias,
                                                   float* __restrict__ out) {
    __shared__ __bf16 Bt[256][PLP];             // [in-channel K][spatial N]
    __shared__ float  sbias[256];
    const int t = threadIdx.x;
    sbias[t] = bias[t];

    const int colBase = blockIdx.x * 64;
    const int nimg = colBase >> 12;
    const int hw   = colBase & (HW_DIM - 1);

    const int lane = t & 31, w = t >> 5;
    const int rl = lane & 15;
    const int ab = (lane & 16) ? 8 : 0;
    const int bb = (lane & 16) ? 16 : 0;

    __builtin_prefetch(Pw + (size_t)(w * 32 + rl) * C_DIM, 0, 2);

    if (t < 32) {
        tdm_load_2d(lds_off(&Bt[0][0]), xb + colBase, 64, 256, M_TOT, 4, 3);
        __builtin_amdgcn_s_wait_tensorcnt(0);
    }
    __syncthreads();

    v8f acc[2][4] = {};
    for (int kb = 0; kb < C_DIM; kb += 32) {
        v16bf a0, a1;
        const int r0 = w * 32 + rl, r1 = w * 32 + 16 + rl;
#pragma unroll
        for (int j = 0; j < 8; ++j) {
            a0[j]     = Pw[(size_t)r0 * C_DIM + kb + ab + j];
            a0[j + 8] = Pw[(size_t)r0 * C_DIM + kb + 16 + ab + j];
            a1[j]     = Pw[(size_t)r1 * C_DIM + kb + ab + j];
            a1[j + 8] = Pw[(size_t)r1 * C_DIM + kb + 16 + ab + j];
        }
#pragma unroll
        for (int nt = 0; nt < 4; ++nt) {
            v16bf b;
#pragma unroll
            for (int j = 0; j < 8; ++j) {
                b[j]     = Bt[kb + ab + j][nt * 16 + rl];
                b[j + 8] = Bt[kb + 16 + ab + j][nt * 16 + rl];
            }
            acc[0][nt] = __builtin_amdgcn_wmma_f32_16x16x32_bf16(false, a0, false, b,
                                                                 (short)0, acc[0][nt], false, false);
            acc[1][nt] = __builtin_amdgcn_wmma_f32_16x16x32_bf16(false, a1, false, b,
                                                                 (short)0, acc[1][nt], false, false);
        }
    }

    const int rowoff = (lane & 16) ? 8 : 0;
#pragma unroll
    for (int m = 0; m < 2; ++m) {
#pragma unroll
        for (int nt = 0; nt < 4; ++nt) {
#pragma unroll
            for (int i = 0; i < 8; ++i) {
                const int row = w * 32 + m * 16 + rowoff + i;     // out channel
                const int col = hw + nt * 16 + rl;                // spatial in image
                out[((size_t)(nimg * C_DIM + row)) * HW_DIM + col] =
                    acc[m][nt][i] + sbias[row];
            }
        }
    }
}

// ---------------------------------------------------------------------------
extern "C" void kernel_launch(void* const* d_in, const int* in_sizes, int n_in,
                              void* d_out, int out_size, void* d_ws, size_t ws_size,
                              hipStream_t stream) {
    const float* x      = (const float*)d_in[0];
    const float* weight = (const float*)d_in[1];
    const float* bias   = (const float*)d_in[2];
    float* out = (float*)d_out;

    char*   ws   = (char*)d_ws;
    float*  mu   = (float*)(ws);
    float*  rstd = (float*)(ws + 1024);
    float*  xxt  = (float*)(ws + 4096);
    __bf16* Pw   = (__bf16*)(ws + 4096 + C_DIM * C_DIM * sizeof(float));
    __bf16* xb   = (__bf16*)(ws + (1u << 20));   // 64 MiB normalized bf16 matrix

    hipMemsetAsync(xxt, 0, C_DIM * C_DIM * sizeof(float), stream);

    stats_kernel<<<C_DIM, 256, 0, stream>>>(x, mu, rstd);
    normconv_kernel<<<dim3(C_DIM, N_IMG), 256, 0, stream>>>(x, mu, rstd, xb);

    const int gram_lds = 2 * (2 * 64 * GLP) * (int)sizeof(__bf16);  // 135168 B
    hipFuncSetAttribute((const void*)gram_kernel,
                        hipFuncAttributeMaxDynamicSharedMemorySize, gram_lds);
    gram_kernel<<<dim3(16, N_IMG), 256, gram_lds, stream>>>(xb, xxt);

    const int eig_lds = C_DIM * C_DIM * (int)sizeof(float);         // 256 KiB
    hipFuncSetAttribute((const void*)eigen_kernel,
                        hipFuncAttributeMaxDynamicSharedMemorySize, eig_lds);
    eigen_kernel<<<1, 256, eig_lds, stream>>>(xxt, weight, Pw);

    proj_kernel<<<M_TOT / 64, 256, 0, stream>>>(xb, Pw, bias, out);
}
